// USAD_LSTM_63471026700373
// MI455X (gfx1250) — compile-verified, matrix operand order checked by reading, not run on hardware
//
#include <hip/hip_runtime.h>

// ---------------------------------------------------------------------------
// USAD LSTM autoencoder pipeline for MI455X (gfx1250, wave32, WMMA).
//
// Roofline: ~19.6 GFLOP vs ~256MB HBM traffic (~11us @ 23.3 TB/s) => compute
// bound on the matrix path.  All 10 LSTM cells (enc x2 passes, dec x3 passes)
// advance in lockstep per timestep so no intermediate sequence ever touches
// HBM.  Gate matmuls run on v_wmma_f32_16x16x32_f16 with f32 accumulation;
// weights are pre-swizzled into WMMA B-operand lane layout in LDS.
//
// v3: gate WMMAs use the free inline-0 C operand with bias folded into the
// elementwise pass (bias scalars hoisted into registers); v_tanh_f32 fast
// path when the builtin exists; unpredicated 16x16 h8 staging (garbage cols
// hit zero-padded w_hr rows); immediate-offset global addressing.
// ---------------------------------------------------------------------------

typedef __attribute__((ext_vector_type(16))) _Float16 v16h;
typedef __attribute__((ext_vector_type(8)))  _Float16 v8h;
typedef __attribute__((ext_vector_type(8)))  float    v8f;
typedef __attribute__((ext_vector_type(4)))  float    v4f;

#define TWIN          16   // window length
#define NFEAT         64   // feature dim / decoder proj dim
#define NLAT          16   // encoder hidden
#define NHID          8    // decoder hidden
#define WAVES         4    // waves per block
#define SEQ_PER_WAVE  16   // WMMA M-tile
#define TILE_F16      512  // one 32x16 B-operand tile (f16 elems)

// Per-wave LDS state slots (units of f16).
#define SLOT_E1   0        // 16x16
#define SLOT_E2   256
#define SLOT_E3   512
#define SLOT_E4   768
#define SLOT_D1A  1024     // 16x64 each from here
#define SLOT_D1B  2048
#define SLOT_D2A  3072
#define SLOT_D2B  4096
#define SLOT_D3A  5120
#define SLOT_D3B  6144
#define WAVE_LDS_F16 7168

// d_in[1+i] order assuming jax tree_leaves flattening (dict keys sorted,
// None (enc w_hr) dropped): dec1[0], dec1[1], dec2[0], dec2[1], enc[0], enc[1]
// with per-layer leaves b, w_hh, w_hr, w_ih.
enum {
  D1L0_B = 0, D1L0_WHH, D1L0_WHR, D1L0_WIH,
  D1L1_B,     D1L1_WHH, D1L1_WHR, D1L1_WIH,
  D2L0_B,     D2L0_WHH, D2L0_WHR, D2L0_WIH,
  D2L1_B,     D2L1_WHH, D2L1_WHR, D2L1_WIH,
  ENC0_B, ENC0_WHH, ENC0_WIH,
  ENC1_B, ENC1_WHH, ENC1_WIH,
  N_WPTRS
};

// LDS B-tile directory (tile index units of TILE_F16).
enum {
  T_ENC1_WIH = 0,  T_ENC1_WHH = 8,
  T_ENC2_WIH = 12, T_ENC2_WHH = 16,
  T_D1L1_WIH = 20, T_D1L1_WHH = 22, T_D1L1_WHR = 26,
  T_D1L2_WIH = 30, T_D1L2_WHH = 34, T_D1L2_WHR = 38,
  T_D2L1_WIH = 42, T_D2L1_WHH = 44, T_D2L1_WHR = 48,
  T_D2L2_WIH = 52, T_D2L2_WHH = 56, T_D2L2_WHR = 60,
  N_TILES    = 64
};

struct WPtrs { const float* p[N_WPTRS]; };

// ---- small helpers ---------------------------------------------------------

__device__ __forceinline__ v8f zero8() {
  v8f z;
#pragma unroll
  for (int i = 0; i < 8; ++i) z[i] = 0.0f;
  return z;
}
__device__ __forceinline__ v16h zero16h() {
  v16h z;
#pragma unroll
  for (int i = 0; i < 16; ++i) z[i] = (_Float16)0.0f;
  return z;
}
__device__ __forceinline__ v8h zero8h() {
  v8h z;
#pragma unroll
  for (int i = 0; i < 8; ++i) z[i] = (_Float16)0.0f;
  return z;
}

__device__ __forceinline__ void lds_fence() {
  asm volatile("s_wait_dscnt 0" ::: "memory");
}

// gfx1250 has V_TANH_F32; use it if clang exposes the builtin.
#if __has_builtin(__builtin_amdgcn_tanhf)
__device__ __forceinline__ float tanhf_(float x) {
  return __builtin_amdgcn_tanhf(x);
}
__device__ __forceinline__ float sigmoidf_(float x) {
  return __builtin_amdgcn_tanhf(0.5f * x) * 0.5f + 0.5f;
}
#else
__device__ __forceinline__ float tanhf_(float x) {
  float e = __expf(-2.0f * x);
  return (1.0f - e) / (1.0f + e);
}
__device__ __forceinline__ float sigmoidf_(float x) {
  return 1.0f / (1.0f + __expf(-x));
}
#endif

__device__ __forceinline__ v8f wmma_f16(v16h a, v16h b, v8f c) {
  // (neg_a, A, neg_b, B, c_mod, C, reuse_a, reuse_b)
  return __builtin_amdgcn_wmma_f32_16x16x32_f16(false, a, false, b,
                                                (short)0, c, false, false);
}

// B operand (32x16 f16) tile: per lane 16 contiguous f16 in pre-swizzled LDS.
__device__ __forceinline__ v16h ldsB(const _Float16* wt, int tile, int lane) {
  const _Float16* p = wt + tile * TILE_F16 + lane * 16;
  v8h lo = *(const v8h*)(p);
  v8h hi = *(const v8h*)(p + 8);
  v16h r;
#pragma unroll
  for (int i = 0; i < 8; ++i) { r[i] = lo[i]; r[8 + i] = hi[i]; }
  return r;
}

// Load an A operand (16 rows x K cols, row-major f16 in LDS) as (K+31)/32
// chunks in the 16-bit A lane layout:
//   k = ((e>>3)<<4) + ((lane>>4)<<3) + (e&7)   (two runs of 8 contiguous k)
// K is 16 or 64 here, so every guard folds to a compile-time constant.
template <int K>
__device__ __forceinline__ void load_a_lds(v16h* a, const _Float16* src, int lane) {
  constexpr int KC = (K + 31) / 32;
  const int m  = lane & 15;
  const int hl = lane >> 4;
#pragma unroll
  for (int c = 0; c < KC; ++c) {
    v16h av = zero16h();
    const int k0 = c * 32 + hl * 8;
    {
      v8h r = *(const v8h*)(src + m * K + k0);
#pragma unroll
      for (int i = 0; i < 8; ++i) av[i] = r[i];
    }
    if (K >= 32) {
      v8h r = *(const v8h*)(src + m * K + k0 + 16);
#pragma unroll
      for (int i = 0; i < 8; ++i) av[8 + i] = r[i];
    }
    a[c] = av;
  }
}

// Store one C/D tile (f32, C layout: vgpr j -> row j + 8*(lane>=16), col lane&15)
// into row-major f16 LDS with row stride `ncols`, column base ntile*16.
__device__ __forceinline__ void store_c_f16(_Float16* dst, int ncols, int ntile,
                                            const v8f& d, int lane) {
  const int n  = lane & 15;
  const int hl = lane >> 4;
#pragma unroll
  for (int j = 0; j < 8; ++j) {
    const int m = j + hl * 8;
    dst[m * ncols + ntile * 16 + n] = (_Float16)d[j];
  }
}

// x_t A-tile straight from global f32.  `row` already folds seq0, lane row m,
// the lane half-offset hl*8 and t; loads are base + immediate.
__device__ __forceinline__ void load_x_a(v16h* a, const float* __restrict__ row) {
#pragma unroll
  for (int c = 0; c < 2; ++c) {
    v4f r0 = *(const v4f*)(row + c * 32);
    v4f r1 = *(const v4f*)(row + c * 32 + 4);
    v4f r2 = *(const v4f*)(row + c * 32 + 16);
    v4f r3 = *(const v4f*)(row + c * 32 + 20);
    v16h av;
#pragma unroll
    for (int i = 0; i < 4; ++i) {
      av[i]      = (_Float16)r0[i];
      av[4 + i]  = (_Float16)r1[i];
      av[8 + i]  = (_Float16)r2[i];
      av[12 + i] = (_Float16)r3[i];
    }
    a[c] = av;
  }
  __builtin_prefetch(row + TWIN * NFEAT / TWIN, 0, 1);  // next t (speculative)
}

// `base` folds out-ptr, seq0, t, lane row-half and column n; all 32 stores
// are base + immediate offset.
__device__ __forceinline__ void store_out(float* __restrict__ base, const v8f* dp) {
#pragma unroll
  for (int j = 0; j < 8; ++j)
#pragma unroll
    for (int nt = 0; nt < 4; ++nt)
      base[j * (TWIN * NFEAT) + nt * 16] = dp[nt][j];
}

// Cooperative: swizzle weight matrix W[outN][inK] (row major f32) into
// B-operand tiles: tile = nt*kc + kt, lane l holds col n = nt*16 + (l&15),
// element e holds k = kt*32 + ((e>>3)<<4) + ((l>>4)<<3) + (e&7); zero-padded.
__device__ void fill_btiles(_Float16* dst, const float* __restrict__ W,
                            int outN, int inK) {
  const int kc    = (inK + 31) >> 5;
  const int total = (outN >> 4) * kc * TILE_F16;
  for (int idx = threadIdx.x; idx < total; idx += blockDim.x) {
    const int e    = idx & 15;
    const int lane = (idx >> 4) & 31;
    const int tile = idx >> 9;
    const int kt   = tile % kc;
    const int nt   = tile / kc;
    const int n = nt * 16 + (lane & 15);
    const int k = kt * 32 + ((e >> 3) << 4) + ((lane >> 4) << 3) + (e & 7);
    dst[idx] = (_Float16)((k < inK) ? W[n * inK + k] : 0.0f);
  }
}

// ---- LSTM cells (recurrent h lives in LDS slots) ---------------------------

// Encoder cell: gates [16,64] = 4 N-tiles (i,f,g,o each exactly one tile).
// Gate WMMAs accumulate from inline C=0; bias bv[4] (per-lane registers) is
// folded into the nonlinearity arguments.
template <int IN_KC>
__device__ __forceinline__ void enc_cell(const v16h* xa, v8f& cst,
                                         _Float16* hslot,
                                         const _Float16* wt, int wih_t, int whh_t,
                                         const float* bv, int lane) {
  v16h ha;
  load_a_lds<NLAT>(&ha, hslot, lane);            // h_{t-1} as A operand
  v8f d[4];
#pragma unroll
  for (int nt = 0; nt < 4; ++nt) {
    v8f acc = zero8();
#pragma unroll
    for (int c = 0; c < IN_KC; ++c)
      acc = wmma_f16(xa[c], ldsB(wt, wih_t + nt * IN_KC + c, lane), acc);
    acc = wmma_f16(ha, ldsB(wt, whh_t + nt, lane), acc);
    d[nt] = acc;
  }
  v8f hnew;
#pragma unroll
  for (int j = 0; j < 8; ++j) {
    const float ig = sigmoidf_(d[0][j] + bv[0]);
    const float fg = sigmoidf_(d[1][j] + bv[1]);
    const float gg = tanhf_(d[2][j] + bv[2]);
    const float og = sigmoidf_(d[3][j] + bv[3]);
    const float cc = fg * cst[j] + ig * gg;
    cst[j]  = cc;
    hnew[j] = og * tanhf_(cc);
  }
  store_c_f16(hslot, NLAT, 0, hnew, lane);       // C layout -> row-major f16
  lds_fence();
}

// Decoder cell with projection: gates [16,32] = 2 N-tiles
// (tile0 = i|f cols, tile1 = g|o cols), then h8 @ w_hr^T -> [16,64].
// h slot: 16x64 f16 row-major (projected h; also this cell's output).
template <int IN_KC>
__device__ __forceinline__ void dec_cell(const v16h* xa, v8f& cst,
                                         _Float16* hslot,
                                         const _Float16* wt, int wih_t, int whh_t,
                                         int whr_t, const float* bv,
                                         int lane, v8f* dproj) {
  v16h hh[2];
  load_a_lds<NFEAT>(hh, hslot, lane);            // projected h_{t-1}
  v8f d[2];
#pragma unroll
  for (int nt = 0; nt < 2; ++nt) {
    v8f acc = zero8();
#pragma unroll
    for (int c = 0; c < IN_KC; ++c)
      acc = wmma_f16(xa[c], ldsB(wt, wih_t + nt * IN_KC + c, lane), acc);
#pragma unroll
    for (int c = 0; c < 2; ++c)
      acc = wmma_f16(hh[c], ldsB(wt, whh_t + nt * 2 + c, lane), acc);
    d[nt] = acc;
  }
  // Bias before the shuffle so each lane adds its own gate's bias, then pair
  // (i,f) and (g,o) across the n<8 / n>=8 lane halves.
  v8f h8;
#pragma unroll
  for (int j = 0; j < 8; ++j) {
    const float a0 = d[0][j] + bv[0];
    const float a1 = d[1][j] + bv[1];
    const float b0 = __shfl_xor(a0, 8, 32);
    const float b1 = __shfl_xor(a1, 8, 32);
    const float ig = sigmoidf_(a0);   // valid on lanes n<8
    const float fg = sigmoidf_(b0);
    const float gg = tanhf_(a1);
    const float og = sigmoidf_(b1);
    const float cc = fg * cst[j] + ig * gg;
    cst[j] = cc;
    h8[j]  = og * tanhf_(cc);
  }
  // Unpredicated 16x16 staging: cols 8..15 hold bounded (sigmoid/tanh-limited)
  // garbage that multiplies the zero-padded k>=8 rows of the w_hr B tiles.
  store_c_f16(hslot, NLAT, 0, h8, lane);
  lds_fence();
  v16h a8;
  load_a_lds<NLAT>(&a8, hslot, lane);
#pragma unroll
  for (int nt = 0; nt < 4; ++nt)
    dproj[nt] = wmma_f16(a8, ldsB(wt, whr_t + nt, lane), zero8());
#pragma unroll
  for (int nt = 0; nt < 4; ++nt)
    store_c_f16(hslot, NFEAT, nt, dproj[nt], lane);  // new projected h
  lds_fence();
}

// ---- kernel ---------------------------------------------------------------

__global__ void __launch_bounds__(WAVES * 32)
usad_lstm_kernel(const float* __restrict__ g, WPtrs w,
                 float* __restrict__ out, long nseq) {
  extern __shared__ char smem[];
  _Float16* wt     = (_Float16*)smem;                                   // 64 KB
  _Float16* stbase = (_Float16*)(smem + N_TILES * TILE_F16 * 2);        // 14KB/wave

  // One-time weight swizzle into B-operand layout (block-cooperative).
  fill_btiles(wt + T_ENC1_WIH * TILE_F16, w.p[ENC0_WIH], 64, 64);
  fill_btiles(wt + T_ENC1_WHH * TILE_F16, w.p[ENC0_WHH], 64, 16);
  fill_btiles(wt + T_ENC2_WIH * TILE_F16, w.p[ENC1_WIH], 64, 16);
  fill_btiles(wt + T_ENC2_WHH * TILE_F16, w.p[ENC1_WHH], 64, 16);
  fill_btiles(wt + T_D1L1_WIH * TILE_F16, w.p[D1L0_WIH], 32, 16);
  fill_btiles(wt + T_D1L1_WHH * TILE_F16, w.p[D1L0_WHH], 32, 64);
  fill_btiles(wt + T_D1L1_WHR * TILE_F16, w.p[D1L0_WHR], 64, 8);
  fill_btiles(wt + T_D1L2_WIH * TILE_F16, w.p[D1L1_WIH], 32, 64);
  fill_btiles(wt + T_D1L2_WHH * TILE_F16, w.p[D1L1_WHH], 32, 64);
  fill_btiles(wt + T_D1L2_WHR * TILE_F16, w.p[D1L1_WHR], 64, 8);
  fill_btiles(wt + T_D2L1_WIH * TILE_F16, w.p[D2L0_WIH], 32, 16);
  fill_btiles(wt + T_D2L1_WHH * TILE_F16, w.p[D2L0_WHH], 32, 64);
  fill_btiles(wt + T_D2L1_WHR * TILE_F16, w.p[D2L0_WHR], 64, 8);
  fill_btiles(wt + T_D2L2_WIH * TILE_F16, w.p[D2L1_WIH], 32, 64);
  fill_btiles(wt + T_D2L2_WHH * TILE_F16, w.p[D2L1_WHH], 32, 64);
  fill_btiles(wt + T_D2L2_WHR * TILE_F16, w.p[D2L1_WHR], 64, 8);

  const int  lane = threadIdx.x & 31;
  const int  wave = threadIdx.x >> 5;
  const int  nsel = lane & 15;
  const int  hl   = lane >> 4;
  _Float16*  st   = stbase + wave * WAVE_LDS_F16;
  const long seq0 = ((long)blockIdx.x * WAVES + wave) * SEQ_PER_WAVE;
  const long OFF  = nseq * TWIN * NFEAT;

  // Zero this wave's h-state slots (h_0 = 0).
  {
    const v8h z = zero8h();
    for (int i = lane * 8; i < WAVE_LDS_F16; i += 32 * 8)
      *(v8h*)(st + i) = z;
  }
  __syncthreads();

  // Per-lane gate biases, hoisted into registers for the whole t loop.
  float be1[4], be2[4], bd1a[2], bd1b[2], bd2a[2], bd2b[2];
#pragma unroll
  for (int nt = 0; nt < 4; ++nt) {
    be1[nt] = w.p[ENC0_B][nt * 16 + nsel];
    be2[nt] = w.p[ENC1_B][nt * 16 + nsel];
  }
#pragma unroll
  for (int nt = 0; nt < 2; ++nt) {
    bd1a[nt] = w.p[D1L0_B][nt * 16 + nsel];
    bd1b[nt] = w.p[D1L1_B][nt * 16 + nsel];
    bd2a[nt] = w.p[D2L0_B][nt * 16 + nsel];
    bd2b[nt] = w.p[D2L1_B][nt * 16 + nsel];
  }

  // f32 cell states (c_0 = 0) stay in VGPRs: 4 enc + 6 dec tiles.
  v8f c_e1 = zero8(), c_e2 = zero8(), c_e3 = zero8(), c_e4 = zero8();
  v8f c_d1a = zero8(), c_d1b = zero8(), c_d2a = zero8(),
      c_d2b = zero8(), c_d3a = zero8(), c_d3b = zero8();

  // Hoisted address bases: all hot-loop memory ops are base + immediate.
  const float* xrow0 = g + (seq0 + nsel) * (TWIN * NFEAT) + hl * 8;
  float*       orow0 = out + seq0 * (TWIN * NFEAT)
                         + (long)hl * 8 * (TWIN * NFEAT) + nsel;

  v8f dp[4];
  v16h xin[2];
#pragma unroll 1
  for (int t = 0; t < TWIN; ++t) {
    const float* xrow = xrow0 + t * NFEAT;
    float*       orow = orow0 + t * NFEAT;

    // encoder pass 1: x_t -> out_n_t (slot E2)
    load_x_a(xin, xrow);
    enc_cell<2>(xin, c_e1, st + SLOT_E1, wt, T_ENC1_WIH, T_ENC1_WHH, be1, lane);
    load_a_lds<NLAT>(xin, st + SLOT_E1, lane);
    enc_cell<1>(xin, c_e2, st + SLOT_E2, wt, T_ENC2_WIH, T_ENC2_WHH, be2, lane);

    v16h on_a;                                    // out_n_t as A operand
    load_a_lds<NLAT>(&on_a, st + SLOT_E2, lane);

    // dec1(out_n) -> ae1_t (slot D1B)
    dec_cell<1>(&on_a, c_d1a, st + SLOT_D1A, wt, T_D1L1_WIH, T_D1L1_WHH,
                T_D1L1_WHR, bd1a, lane, dp);
    load_a_lds<NFEAT>(xin, st + SLOT_D1A, lane);
    dec_cell<2>(xin, c_d1b, st + SLOT_D1B, wt, T_D1L2_WIH, T_D1L2_WHH,
                T_D1L2_WHR, bd1b, lane, dp);
    store_out(orow, dp);

    // dec2(out_n) -> ae2_t
    dec_cell<1>(&on_a, c_d2a, st + SLOT_D2A, wt, T_D2L1_WIH, T_D2L1_WHH,
                T_D2L1_WHR, bd2a, lane, dp);
    load_a_lds<NFEAT>(xin, st + SLOT_D2A, lane);
    dec_cell<2>(xin, c_d2b, st + SLOT_D2B, wt, T_D2L2_WIH, T_D2L2_WHH,
                T_D2L2_WHR, bd2b, lane, dp);
    store_out(orow + OFF, dp);

    // encoder pass 2 on ae1_t (read from slot D1B)
    load_a_lds<NFEAT>(xin, st + SLOT_D1B, lane);
    enc_cell<2>(xin, c_e3, st + SLOT_E3, wt, T_ENC1_WIH, T_ENC1_WHH, be1, lane);
    load_a_lds<NLAT>(xin, st + SLOT_E3, lane);
    enc_cell<1>(xin, c_e4, st + SLOT_E4, wt, T_ENC2_WIH, T_ENC2_WHH, be2, lane);

    // dec2 pass 2 -> ae2ae1_t
    load_a_lds<NLAT>(xin, st + SLOT_E4, lane);
    dec_cell<1>(xin, c_d3a, st + SLOT_D3A, wt, T_D2L1_WIH, T_D2L1_WHH,
                T_D2L1_WHR, bd2a, lane, dp);
    load_a_lds<NFEAT>(xin, st + SLOT_D3A, lane);
    dec_cell<2>(xin, c_d3b, st + SLOT_D3B, wt, T_D2L2_WIH, T_D2L2_WHH,
                T_D2L2_WHR, bd2b, lane, dp);
    store_out(orow + 2 * OFF, dp);
  }
}

// ---- host entry ------------------------------------------------------------

extern "C" void kernel_launch(void* const* d_in, const int* in_sizes, int n_in,
                              void* d_out, int out_size, void* d_ws, size_t ws_size,
                              hipStream_t stream) {
  (void)n_in; (void)out_size; (void)d_ws; (void)ws_size;
  const float* g = (const float*)d_in[0];
  WPtrs w;
  for (int i = 0; i < N_WPTRS; ++i) w.p[i] = (const float*)d_in[1 + i];

  const long nseq   = (long)in_sizes[0] / (TWIN * NFEAT);   // 16384
  const int  blocks = (int)(nseq / (WAVES * SEQ_PER_WAVE)); // 256
  const size_t smem = (size_t)N_TILES * TILE_F16 * 2       // B tiles (f16)
                    + (size_t)WAVES * WAVE_LDS_F16 * 2;    // per-wave h slots

  usad_lstm_kernel<<<blocks, WAVES * 32, smem, stream>>>(g, w, (float*)d_out, nseq);
}
